// SeqClassifier_29368986370088
// MI455X (gfx1250) — matrix operand order repaired and checked by reading, not compiled, not run on hardware
//
#include <hip/hip_runtime.h>
#include <cmath>

// ---------------- constants (match reference) ----------------
#define LL     4096
#define KNN    16
#define DSEQ   768
#define HID    256
#define INDIM  1540
#define KPAD   1568   // 1540 padded to 49*32 for K%32==0 WMMA stepping
#define NGCN   8

typedef unsigned short bf16_t;
typedef __attribute__((ext_vector_type(16))) __bf16 v16bf;
typedef __attribute__((ext_vector_type(8)))  float  v8f;

// ---------------- device helpers ----------------
static __device__ __forceinline__ bf16_t f2bf(float f) {
  unsigned u = __float_as_uint(f);
  u += 0x7FFFu + ((u >> 16) & 1u);        // round-to-nearest-even
  return (bf16_t)(u >> 16);
}

static __device__ __forceinline__ float act_apply(float v, int act) {
  switch (act) {
    case 1: return v > 0.f ? v : 0.f;                                   // relu
    case 2: return tanhf(v);                                            // tanh
    case 3: return 1.f / (1.f + __expf(-v));                            // sigmoid
    case 4: return 0.5f * v * (1.f + erff(v * 0.7071067811865476f));    // exact gelu
  }
  return v;
}

// blockDim.x == 256 assumed
static __device__ __forceinline__ void block_reduce_add2(float* s1, float* s2) {
  int t = threadIdx.x;
  for (int off = 128; off > 0; off >>= 1) {
    __syncthreads();
    if (t < off) { s1[t] += s1[t + off]; s2[t] += s2[t + off]; }
  }
  __syncthreads();
}

static __device__ __forceinline__ void ln4(const float* x, const float* g,
                                           const float* b, float* y) {
  float m = 0.25f * (x[0] + x[1] + x[2] + x[3]);
  float v = 0.f;
#pragma unroll
  for (int i = 0; i < 4; ++i) { float d = x[i] - m; v += d * d; }
  v *= 0.25f;
  float r = rsqrtf(v + 1e-5f);
#pragma unroll
  for (int i = 0; i < 4; ++i) y[i] = (x[i] - m) * r * g[i] + b[i];
}

// ---------------- WMMA GEMM: C[M,N] = act(A[M,K]@B[K,N] + bias) ----------------
// A: bf16 row-major (lda=K), B: bf16 row-major (ldb=N).
// Register-blocked: one wave computes a 32x32 output tile (2x2 WMMA tiles),
// so each A/B fragment is reused twice -> 4 v_wmma per 4 b128 A loads + 32 16-bit B loads per k-step.
// Requires M%32==0, N%32==0, K%32==0 (true for all GEMMs in this net).
// Operand layouts per CDNA5 ISA 7.12.2:
//   A frag (16x32 bf16): lane holds row m=lane&15; k = 8*(lane>>4) + e (+8 for e>=8)
//   B frag (32x16 bf16): lane holds column n=lane&15; k = 8*(lane>>4) + e (+8 for e>=8)
//   C/D  (16x16 f32):    elem j -> m = j + 8*(lane>>4), n = lane&15
__global__ void gemm_bf16_wmma(const bf16_t* __restrict__ A,
                               const bf16_t* __restrict__ B,
                               const float* __restrict__ bias,
                               float* __restrict__ C,
                               int M, int N, int Kd, int act) {
  int wave   = blockIdx.x * (blockDim.x >> 5) + (threadIdx.x >> 5);
  int tilesN = N >> 5;
  int tm = wave / tilesN, tn = wave - tm * tilesN;
  if (tm * 32 >= M) return;
  int lane = threadIdx.x & 31;
  int hi   = lane >> 4;                          // 0 or 1
  const bf16_t* arow0 = A + (size_t)(tm * 32 + (lane & 15)) * Kd + hi * 8;
  const bf16_t* arow1 = arow0 + (size_t)16 * Kd;
  union Frag { uint4 q[2]; v16bf v; };
  v8f acc00 = {0.f, 0.f, 0.f, 0.f, 0.f, 0.f, 0.f, 0.f};
  v8f acc01 = acc00, acc10 = acc00, acc11 = acc00;
  for (int k0 = 0; k0 < Kd; k0 += 32) {
    Frag a0, a1, b0, b1;
    a0.q[0] = *(const uint4*)(arow0 + k0);        // k = k0 + hi*8 + [0..7]
    a0.q[1] = *(const uint4*)(arow0 + k0 + 16);   // k = k0 + hi*8 + [16..23]
    a1.q[0] = *(const uint4*)(arow1 + k0);
    a1.q[1] = *(const uint4*)(arow1 + k0 + 16);
    const bf16_t* bp = B + (size_t)(k0 + hi * 8) * N + tn * 32 + (lane & 15);
    bf16_t* b0e = (bf16_t*)&b0;
    bf16_t* b1e = (bf16_t*)&b1;
#pragma unroll
    for (int s = 0; s < 8; ++s) {
      b0e[s]     = bp[(size_t)s * N];
      b0e[8 + s] = bp[(size_t)(16 + s) * N];
      b1e[s]     = bp[(size_t)s * N + 16];
      b1e[8 + s] = bp[(size_t)(16 + s) * N + 16];
    }
    if (k0 + 32 < Kd) {
      __builtin_prefetch(arow0 + k0 + 32, 0, 1);
      __builtin_prefetch(arow1 + k0 + 32, 0, 1);
    }
    acc00 = __builtin_amdgcn_wmma_f32_16x16x32_bf16(false, a0.v, false, b0.v,
                                                    (short)0, acc00, false, false);
    acc01 = __builtin_amdgcn_wmma_f32_16x16x32_bf16(false, a0.v, false, b1.v,
                                                    (short)0, acc01, false, false);
    acc10 = __builtin_amdgcn_wmma_f32_16x16x32_bf16(false, a1.v, false, b0.v,
                                                    (short)0, acc10, false, false);
    acc11 = __builtin_amdgcn_wmma_f32_16x16x32_bf16(false, a1.v, false, b1.v,
                                                    (short)0, acc11, false, false);
  }
  int nc0 = tn * 32 + (lane & 15);
  int nc1 = nc0 + 16;
  float bv0 = bias ? bias[nc0] : 0.f;
  float bv1 = bias ? bias[nc1] : 0.f;
#pragma unroll
  for (int j = 0; j < 8; ++j) {
    int mA = tm * 32 + j + 8 * hi;
    int mB = mA + 16;
    C[(size_t)mA * N + nc0] = act_apply(acc00[j] + bv0, act);
    C[(size_t)mA * N + nc1] = act_apply(acc01[j] + bv1, act);
    C[(size_t)mB * N + nc0] = act_apply(acc10[j] + bv0, act);
    C[(size_t)mB * N + nc1] = act_apply(acc11[j] + bv1, act);
  }
}

// ---------------- conversion kernels ----------------
__global__ void cvt_bf16_kernel(const float* __restrict__ s, bf16_t* __restrict__ d, int n) {
  int i = blockIdx.x * blockDim.x + threadIdx.x;
  if (i < n) d[i] = f2bf(s[i]);
}

__global__ void cvt_fc_pad_kernel(const float* __restrict__ s, bf16_t* __restrict__ d) {
  int i = blockIdx.x * blockDim.x + threadIdx.x;       // over KPAD*HID
  if (i >= KPAD * HID) return;
  int r = i >> 8;                                      // row (k), col = i&255
  d[i] = (r < INDIM) ? f2bf(s[i]) : (bf16_t)0;
}

// s: [N][K] (e.g. pointwise conv [cout][cin]) -> d: [K][N]
__global__ void cvt_transpose_kernel(const float* __restrict__ s, bf16_t* __restrict__ d,
                                     int NN, int KK) {
  int i = blockIdx.x * blockDim.x + threadIdx.x;
  if (i >= NN * KK) return;
  int k = i / NN, n = i - k * NN;
  d[i] = f2bf(s[(size_t)n * KK + k]);
}

// ---------------- shape encoder (d=4, NHEAD=2, hd=2) ----------------
__global__ void shape_ln_in_kernel(const float* __restrict__ x, const float* g,
                                   const float* b, float* __restrict__ y) {
  int r = blockIdx.x * blockDim.x + threadIdx.x;
  if (r >= LL) return;
  float xi[4], yo[4];
#pragma unroll
  for (int i = 0; i < 4; ++i) xi[i] = x[r * 4 + i];
  ln4(xi, g, b, yo);
#pragma unroll
  for (int i = 0; i < 4; ++i) y[r * 4 + i] = yo[i];
}

__global__ void shape_qkv_kernel(const float* __restrict__ x,
                                 const float* Wq, const float* bq,
                                 const float* Wk, const float* bk,
                                 const float* Wv, const float* bv,
                                 float* __restrict__ q, float* __restrict__ k,
                                 float* __restrict__ v) {
  int r = blockIdx.x * blockDim.x + threadIdx.x;
  if (r >= LL) return;
  float xi[4];
#pragma unroll
  for (int i = 0; i < 4; ++i) xi[i] = x[r * 4 + i];
#pragma unroll
  for (int i = 0; i < 4; ++i) {
    float aq = bq[i], ak = bk[i], av = bv[i];
#pragma unroll
    for (int j = 0; j < 4; ++j) {
      aq += xi[j] * Wq[j * 4 + i];
      ak += xi[j] * Wk[j * 4 + i];
      av += xi[j] * Wv[j * 4 + i];
    }
    q[r * 4 + i] = aq; k[r * 4 + i] = ak; v[r * 4 + i] = av;
  }
}

__global__ void shape_attn_kernel(const float* __restrict__ q, const float* __restrict__ k,
                                  const float* __restrict__ v, float* __restrict__ o) {
  int tid = blockIdx.x * blockDim.x + threadIdx.x;     // 2*LL threads
  if (tid >= 2 * LL) return;
  int h = tid / LL, i = tid - h * LL;
  float q0 = q[i * 4 + 2 * h], q1 = q[i * 4 + 2 * h + 1];
  const float scale = 0.7071067811865476f;             // 1/sqrt(hd=2)
  float m = -3.4e38f, s = 0.f, o0 = 0.f, o1 = 0.f;
  for (int j = 0; j < LL; ++j) {
    float sc = (q0 * k[j * 4 + 2 * h] + q1 * k[j * 4 + 2 * h + 1]) * scale;
    float mn = fmaxf(m, sc);
    float corr = __expf(m - mn);
    float p = __expf(sc - mn);
    s  = s * corr + p;
    o0 = o0 * corr + p * v[j * 4 + 2 * h];
    o1 = o1 * corr + p * v[j * 4 + 2 * h + 1];
    m = mn;
  }
  float inv = 1.f / s;
  o[i * 4 + 2 * h]     = o0 * inv;
  o[i * 4 + 2 * h + 1] = o1 * inv;
}

__global__ void shape_proj_ln_kernel(float* __restrict__ x, const float* __restrict__ o,
                                     const float* Wo, const float* bo,
                                     const float* g, const float* b) {
  int r = blockIdx.x * blockDim.x + threadIdx.x;
  if (r >= LL) return;
  float xi[4], oi[4], y[4], yo[4];
#pragma unroll
  for (int i = 0; i < 4; ++i) { xi[i] = x[r * 4 + i]; oi[i] = o[r * 4 + i]; }
#pragma unroll
  for (int i = 0; i < 4; ++i) {
    float a = bo[i];
#pragma unroll
    for (int j = 0; j < 4; ++j) a += oi[j] * Wo[j * 4 + i];
    y[i] = xi[i] + a;
  }
  ln4(y, g, b, yo);
#pragma unroll
  for (int i = 0; i < 4; ++i) x[r * 4 + i] = yo[i];
}

__global__ void shape_ff_ln_kernel(float* __restrict__ x,
                                   const float* W1, const float* b1,
                                   const float* W2, const float* b2,
                                   const float* g, const float* b) {
  int r = blockIdx.x * blockDim.x + threadIdx.x;
  if (r >= LL) return;
  float xi[4], y[4], yo[4];
#pragma unroll
  for (int i = 0; i < 4; ++i) { xi[i] = x[r * 4 + i]; y[i] = b2[i]; }
  for (int t = 0; t < 64; ++t) {
    float h = b1[t];
#pragma unroll
    for (int j = 0; j < 4; ++j) h += xi[j] * W1[j * 64 + t];
    h = act_apply(h, 4);
#pragma unroll
    for (int i = 0; i < 4; ++i) y[i] += h * W2[t * 4 + i];
  }
#pragma unroll
  for (int i = 0; i < 4; ++i) y[i] += xi[i];
  ln4(y, g, b, yo);
#pragma unroll
  for (int i = 0; i < 4; ++i) x[r * 4 + i] = yo[i];
}

__global__ void shape_fnn_ln_kernel(const float* __restrict__ x,
                                    const float* W1, const float* b1,
                                    const float* W2, const float* b2,
                                    const float* g, const float* b,
                                    float* __restrict__ out) {
  int r = blockIdx.x * blockDim.x + threadIdx.x;
  if (r >= LL) return;
  float xi[4], h[4], y[4], yo[4];
#pragma unroll
  for (int i = 0; i < 4; ++i) xi[i] = x[r * 4 + i];
#pragma unroll
  for (int i = 0; i < 4; ++i) {
    float a = b1[i];
#pragma unroll
    for (int j = 0; j < 4; ++j) a += xi[j] * W1[j * 4 + i];
    h[i] = act_apply(a, 4);
  }
#pragma unroll
  for (int i = 0; i < 4; ++i) {
    float a = b2[i];
#pragma unroll
    for (int j = 0; j < 4; ++j) a += h[j] * W2[j * 4 + i];
    y[i] = xi[i] + a;
  }
  ln4(y, g, b, yo);
#pragma unroll
  for (int i = 0; i < 4; ++i) out[r * 4 + i] = yo[i];
}

// ---------------- align + concat -> x bf16 [LL][KPAD] ----------------
__global__ void align_concat_kernel(const float* __restrict__ seq, const float* __restrict__ nt,
                                    const float* __restrict__ shp,
                                    const float* sg, const float* sb,
                                    const float* ng, const float* nb,
                                    const float* pg, const float* pb,
                                    bf16_t* __restrict__ xbf) {
  int r = blockIdx.x, t = threadIdx.x;
  __shared__ float s1[256], s2[256];
  // seq LN -> cols [0,768)
  {
    const float* row = seq + (size_t)r * DSEQ;
    float a = 0.f, q = 0.f;
    for (int c = t; c < DSEQ; c += 256) { float v = row[c]; a += v; q += v * v; }
    s1[t] = a; s2[t] = q;
    block_reduce_add2(s1, s2);
    float mean = s1[0] * (1.f / DSEQ);
    float var  = s2[0] * (1.f / DSEQ) - mean * mean;
    float rstd = rsqrtf(var + 1e-5f);
    __syncthreads();
    for (int c = t; c < DSEQ; c += 256)
      xbf[(size_t)r * KPAD + c] = f2bf((row[c] - mean) * rstd * sg[c] + sb[c]);
  }
  __syncthreads();
  // nt LN -> cols [768,1536)
  {
    const float* row = nt + (size_t)r * DSEQ;
    float a = 0.f, q = 0.f;
    for (int c = t; c < DSEQ; c += 256) { float v = row[c]; a += v; q += v * v; }
    s1[t] = a; s2[t] = q;
    block_reduce_add2(s1, s2);
    float mean = s1[0] * (1.f / DSEQ);
    float var  = s2[0] * (1.f / DSEQ) - mean * mean;
    float rstd = rsqrtf(var + 1e-5f);
    __syncthreads();
    for (int c = t; c < DSEQ; c += 256)
      xbf[(size_t)r * KPAD + DSEQ + c] = f2bf((row[c] - mean) * rstd * ng[c] + nb[c]);
  }
  // shape LN -> cols [1536,1540); zero pad [1540,1568)
  if (t < 4) {
    float xi[4], yo[4];
#pragma unroll
    for (int i = 0; i < 4; ++i) xi[i] = shp[r * 4 + i];
    ln4(xi, pg, pb, yo);
    xbf[(size_t)r * KPAD + 2 * DSEQ + t] = f2bf(yo[t]);
  } else if (t < 32) {
    xbf[(size_t)r * KPAD + 2 * DSEQ + t] = (bf16_t)0;  // cols 1540..1567
  }
}

// ---------------- GCN helpers ----------------
__global__ void vals0_kernel(const float* __restrict__ knw, float* __restrict__ vals) {
  int r = blockIdx.x * blockDim.x + threadIdx.x;
  if (r >= LL) return;
  float s = 0.f;
#pragma unroll
  for (int j = 0; j < KNN; ++j) s += knw[r * KNN + j];
  float inv = 1.f / (s + 1e-8f);
#pragma unroll
  for (int j = 0; j < KNN; ++j) vals[r * KNN + j] = knw[r * KNN + j] * inv;
}

// S = 0.9*(sum_j vals*h[idx] + 0.1*h[r]) + 0.1*h0[r]; also bf16 copy
__global__ void spmm_support_kernel(const int* __restrict__ idx, const float* __restrict__ vals,
                                    const float* __restrict__ h, const float* __restrict__ h0,
                                    float* __restrict__ S, bf16_t* __restrict__ Sbf) {
  int r = blockIdx.x, c = threadIdx.x;
  __shared__ int   si[KNN];
  __shared__ float sv[KNN];
  if (c < KNN) { si[c] = idx[r * KNN + c]; sv[c] = vals[r * KNN + c]; }
  __syncthreads();
  float acc = 0.1f * h[(size_t)r * HID + c];           // SELF_LOOP_EPS edge
#pragma unroll 4
  for (int j = 0; j < KNN; ++j) acc += sv[j] * h[(size_t)si[j] * HID + c];
  float s = 0.9f * acc + 0.1f * h0[(size_t)r * HID + c];
  S[(size_t)r * HID + c] = s;
  Sbf[(size_t)r * HID + c] = f2bf(s);
}

__global__ void gcn_combine_kernel(float theta, const float* __restrict__ T,
                                   const float* __restrict__ S, float* __restrict__ h, int n) {
  int i = blockIdx.x * blockDim.x + threadIdx.x;
  if (i >= n) return;
  float o = theta * T[i] + (1.f - theta) * S[i] + h[i];
  h[i] = o > 0.f ? o : 0.f;
}

// per-row LN over HID=256; optional residual add, optional bf16 copy
__global__ void ln256_kernel(const float* __restrict__ X, const float* g, const float* b,
                             const float* __restrict__ resid,
                             float* __restrict__ Yf, bf16_t* __restrict__ Ybf) {
  int r = blockIdx.x, t = threadIdx.x;
  __shared__ float s1[256], s2[256];
  float v = X[(size_t)r * HID + t];
  s1[t] = v; s2[t] = v * v;
  block_reduce_add2(s1, s2);
  float mean = s1[0] * (1.f / HID);
  float var  = s2[0] * (1.f / HID) - mean * mean;
  float rstd = rsqrtf(var + 1e-5f);
  float y = (v - mean) * rstd * g[t] + b[t];
  if (resid) y += resid[(size_t)r * HID + t];
  if (Yf)  Yf[(size_t)r * HID + t] = y;
  if (Ybf) Ybf[(size_t)r * HID + t] = f2bf(y);
}

// ---------------- content-masked adjacency ----------------
__global__ void sim_kernel(const float* __restrict__ Q, const float* __restrict__ Km,
                           const int* __restrict__ idx, float* __restrict__ sim) {
  int tid = blockIdx.x * blockDim.x + threadIdx.x;
  if (tid >= LL * KNN) return;
  int l = tid >> 4, j = tid & 15;
  const float* qr = Q + (size_t)l * HID;
  const float* kr = Km + (size_t)idx[l * KNN + j] * HID;
  float s = 0.f;
  for (int c = 0; c < HID; ++c) s += qr[c] * kr[c];
  sim[tid] = s;
}

__global__ void vals1_kernel(const float* __restrict__ sim, const float* __restrict__ knw,
                             const float* __restrict__ mix, float* __restrict__ vals) {
  int r = blockIdx.x * blockDim.x + threadIdx.x;
  if (r >= LL) return;
  float beta = 1.f / (1.f + __expf(-mix[0]));
  float s[KNN], m = -3.4e38f;
#pragma unroll
  for (int j = 0; j < KNN; ++j) { s[j] = sim[r * KNN + j] * 5.0f; m = fmaxf(m, s[j]); } // /TAU
  float tot = 0.f;
#pragma unroll
  for (int j = 0; j < KNN; ++j) { s[j] = __expf(s[j] - m); tot += s[j]; }
  float inv = 1.f / tot;
  float wsum = 0.f;
#pragma unroll
  for (int j = 0; j < KNN; ++j) {
    s[j] = (1.f - beta) * knw[r * KNN + j] + beta * s[j] * inv;
    wsum += s[j];
  }
  float ninv = 1.f / (wsum + 1e-8f);
#pragma unroll
  for (int j = 0; j < KNN; ++j) vals[r * KNN + j] = s[j] * ninv;
}

// ---------------- depthwise conv (cross-correlation, kernel=3) ----------------
__global__ void dwconv_kernel(const float* __restrict__ X, const float* __restrict__ dw,
                              bf16_t* __restrict__ Y, int dil) {
  int tid = blockIdx.x * blockDim.x + threadIdx.x;
  if (tid >= LL * HID) return;
  int l = tid >> 8, c = tid & 255;
  float acc = 0.f;
#pragma unroll
  for (int t = 0; t < 3; ++t) {
    int p = l + (t - 1) * dil;
    if (p >= 0 && p < LL) acc += dw[c * 3 + t] * X[(size_t)p * HID + c];
  }
  Y[tid] = f2bf(acc);
}

// ---------------- readout ----------------
__global__ void readout_att_kernel(const float* __restrict__ rq, const float* __restrict__ H,
                                   float* __restrict__ att) {
  int tid = blockIdx.x * blockDim.x + threadIdx.x;
  if (tid >= 4 * LL) return;
  int q = tid / LL, l = tid - q * LL;
  const float* qr = rq + q * HID;
  const float* hr = H + (size_t)l * HID;
  float s = 0.f;
  for (int c = 0; c < HID; ++c) s += qr[c] * hr[c];
  att[(size_t)q * LL + l] = s * 0.0625f;               // 1/sqrt(256)
}

__global__ void readout_softmax_kernel(float* __restrict__ att) {
  int q = blockIdx.x, t = threadIdx.x;
  float* a = att + (size_t)q * LL;
  __shared__ float sh[256];
  float m = -3.4e38f;
  for (int l = t; l < LL; l += 256) m = fmaxf(m, a[l]);
  sh[t] = m;
  for (int off = 128; off > 0; off >>= 1) { __syncthreads(); if (t < off) sh[t] = fmaxf(sh[t], sh[t + off]); }
  __syncthreads();
  m = sh[0];
  __syncthreads();
  float s = 0.f;
  for (int l = t; l < LL; l += 256) s += __expf(a[l] - m);
  sh[t] = s;
  for (int off = 128; off > 0; off >>= 1) { __syncthreads(); if (t < off) sh[t] += sh[t + off]; }
  __syncthreads();
  float inv = 1.f / sh[0];
  for (int l = t; l < LL; l += 256) a[l] = __expf(a[l] - m) * inv;
}

__global__ void wsum_kernel(const float* __restrict__ att, float* __restrict__ w) {
  int l = blockIdx.x * blockDim.x + threadIdx.x;
  if (l >= LL) return;
  w[l] = 0.25f * (att[l] + att[LL + l] + att[2 * LL + l] + att[3 * LL + l]);
}

__global__ void gvec_kernel(const float* __restrict__ w, const float* __restrict__ H,
                            float* __restrict__ g) {
  int c = threadIdx.x;
  float acc = 0.f;
  for (int l = 0; l < LL; ++l) acc += w[l] * H[(size_t)l * HID + c];
  g[c] = acc;
}

// ---------------- gated nt branch ----------------
__global__ void nt_ln_kernel(const float* __restrict__ nt,
                             const float* pg, const float* pb,
                             const float* gg, const float* gb,
                             bf16_t* __restrict__ outp, bf16_t* __restrict__ outg) {
  int r = blockIdx.x, t = threadIdx.x;
  __shared__ float s1[256], s2[256];
  const float* row = nt + (size_t)r * DSEQ;
  float a = 0.f, q = 0.f;
  for (int c = t; c < DSEQ; c += 256) { float v = row[c]; a += v; q += v * v; }
  s1[t] = a; s2[t] = q;
  block_reduce_add2(s1, s2);
  float mean = s1[0] * (1.f / DSEQ);
  float var  = s2[0] * (1.f / DSEQ) - mean * mean;
  float rstd = rsqrtf(var + 1e-5f);
  __syncthreads();
  for (int c = t; c < DSEQ; c += 256) {
    float xn = (row[c] - mean) * rstd;
    outp[(size_t)r * DSEQ + c] = f2bf(xn * pg[c] + pb[c]);
    outg[(size_t)r * DSEQ + c] = f2bf(xn * gg[c] + gb[c]);
  }
}

__global__ void vmean_kernel(const float* __restrict__ Vs, const float* __restrict__ Vg,
                             float* __restrict__ v) {
  int c = threadIdx.x;
  float acc = 0.f;
  for (int l = 0; l < LL; ++l) acc += Vs[(size_t)l * HID + c] * Vg[(size_t)l * HID + c];
  v[c] = acc * (1.f / LL);
}

// ---------------- MLP head ----------------
__global__ void head_kernel(const float* __restrict__ g, const float* __restrict__ v,
                            const float* W1, const float* b1,
                            const float* W2, const float* b2,
                            const float* W3, const float* b3,
                            float* __restrict__ out) {
  __shared__ float z[512];
  __shared__ float z1[256];
  __shared__ float z2[64];
  int t = threadIdx.x;
  z[t] = g[t]; z[256 + t] = v[t];
  __syncthreads();
  float a = b1[t];
  for (int k = 0; k < 512; ++k) a += z[k] * W1[k * 256 + t];
  z1[t] = a > 0.f ? a : 0.1f * a;
  __syncthreads();
  if (t < 64) {
    float a2 = b2[t];
    for (int k = 0; k < 256; ++k) a2 += z1[k] * W2[k * 64 + t];
    z2[t] = a2 > 0.f ? a2 : 0.1f * a2;
  }
  __syncthreads();
  if (t == 0) {
    float a3 = b3[0];
    for (int k = 0; k < 64; ++k) a3 += z2[k] * W3[k];
    out[0] = a3;
  }
}

// ================= host orchestration =================
extern "C" void kernel_launch(void* const* d_in, const int* in_sizes, int n_in,
                              void* d_out, int out_size, void* d_ws, size_t ws_size,
                              hipStream_t stream) {
  (void)in_sizes; (void)n_in; (void)out_size; (void)ws_size;

  // ----- inputs (setup_inputs dict order, params flattened in insertion order) -----
  const float* seq     = (const float*)d_in[0];
  const float* nt      = (const float*)d_in[1];
  const float* shp     = (const float*)d_in[2];
  const int*   knn_idx = (const int*)d_in[3];
  const float* knn_w   = (const float*)d_in[4];
  const float* se_ln_in_g = (const float*)d_in[5];
  const float* se_ln_in_b = (const float*)d_in[6];
  // layers at 7 + 16*l: Wq bq Wk bk Wv bv Wo bo ln1g ln1b ln2g ln2b W1 b1 W2 b2
  const float* fnn_W1 = (const float*)d_in[39];
  const float* fnn_b1 = (const float*)d_in[40];
  const float* fnn_W2 = (const float*)d_in[41];
  const float* fnn_b2 = (const float*)d_in[42];
  const float* se_ln_out_g = (const float*)d_in[43];
  const float* se_ln_out_b = (const float*)d_in[44];
  const float* align_s_g  = (const float*)d_in[45];
  const float* align_s_b  = (const float*)d_in[46];
  const float* align_nt_g = (const float*)d_in[47];
  const float* align_nt_b = (const float*)d_in[48];
  const float* align_sh_g = (const float*)d_in[49];
  const float* align_sh_b = (const float*)d_in[50];
  const float* fc_in_W = (const float*)d_in[51];
  const float* fc_in_b = (const float*)d_in[52];
  const float* conv_W[NGCN];
  for (int i = 0; i < NGCN; ++i) conv_W[i] = (const float*)d_in[53 + i];
  const float* post_g = (const float*)d_in[61];
  const float* post_b = (const float*)d_in[62];
  const float* cadj_Wq  = (const float*)d_in[63];
  const float* cadj_Wk  = (const float*)d_in[64];
  const float* cadj_mix = (const float*)d_in[65];
  const float* dw1 = (const float*)d_in[66];
  const float* pw1 = (const float*)d_in[67];
  const float* dw2 = (const float*)d_in[68];
  const float* pw2 = (const float*)d_in[69];
  const float* cnn_g = (const float*)d_in[70];
  const float* cnn_b = (const float*)d_in[71];
  const float* readout_q = (const float*)d_in[72];
  const float* ntp_ln_g = (const float*)d_in[73];
  const float* ntp_ln_b = (const float*)d_in[74];
  const float* ntp_W = (const float*)d_in[75];
  const float* ntp_b = (const float*)d_in[76];
  const float* ntg_ln_g = (const float*)d_in[77];
  const float* ntg_ln_b = (const float*)d_in[78];
  const float* ntg_W = (const float*)d_in[79];
  const float* ntg_b = (const float*)d_in[80];
  const float* h1_W = (const float*)d_in[81];
  const float* h1_b = (const float*)d_in[82];
  const float* h2_W = (const float*)d_in[83];
  const float* h2_b = (const float*)d_in[84];
  const float* h3_W = (const float*)d_in[85];
  const float* h3_b = (const float*)d_in[86];

  // ----- workspace layout (~42 MB) -----
  char* base = (char*)d_ws;
  size_t off = 0;
  auto wsa = [&](size_t bytes) -> void* {
    void* p = base + off;
    off = (off + bytes + 255) & ~(size_t)255;
    return p;
  };
  float* P0 = (float*)wsa((size_t)LL * HID * 4);
  float* P1 = (float*)wsa((size_t)LL * HID * 4);
  float* P2 = (float*)wsa((size_t)LL * HID * 4);
  float* P3 = (float*)wsa((size_t)LL * HID * 4);
  float* P4 = (float*)wsa((size_t)LL * HID * 4);
  bf16_t* XBF  = (bf16_t*)wsa((size_t)LL * KPAD * 2);   // reused later for nt-LN bf16
  bf16_t* SBF  = (bf16_t*)wsa((size_t)LL * HID * 2);
  bf16_t* HBF  = (bf16_t*)wsa((size_t)LL * HID * 2);
  bf16_t* WFC  = (bf16_t*)wsa((size_t)KPAD * HID * 2);
  bf16_t* WCV  = (bf16_t*)wsa((size_t)NGCN * HID * HID * 2);
  bf16_t* WQB  = (bf16_t*)wsa((size_t)HID * HID * 2);
  bf16_t* WKB  = (bf16_t*)wsa((size_t)HID * HID * 2);
  bf16_t* PW1T = (bf16_t*)wsa((size_t)HID * HID * 2);
  bf16_t* PW2T = (bf16_t*)wsa((size_t)HID * HID * 2);
  bf16_t* NTPW = (bf16_t*)wsa((size_t)DSEQ * HID * 2);
  bf16_t* NTGW = (bf16_t*)wsa((size_t)DSEQ * HID * 2);
  float* VALS0 = (float*)wsa((size_t)LL * KNN * 4);
  float* VALS1 = (float*)wsa((size_t)LL * KNN * 4);
  float* SIM   = (float*)wsa((size_t)LL * KNN * 4);
  float* SA = (float*)wsa((size_t)LL * 4 * 4);
  float* SB = (float*)wsa((size_t)LL * 4 * 4);
  float* SQ = (float*)wsa((size_t)LL * 4 * 4);
  float* SK = (float*)wsa((size_t)LL * 4 * 4);
  float* SV = (float*)wsa((size_t)LL * 4 * 4);
  float* SO = (float*)wsa((size_t)LL * 4 * 4);
  float* ATT = (float*)wsa((size_t)4 * LL * 4);
  float* WSM = (float*)wsa((size_t)LL * 4);
  float* GV  = (float*)wsa((size_t)HID * 4);
  float* VV  = (float*)wsa((size_t)HID * 4);
  // nt-LN bf16 reuses XBF (free after both GCN passes): 2*LL*768 <= LL*KPAD
  bf16_t* NTP = XBF;
  bf16_t* NTG = XBF + (size_t)LL * DSEQ;

  auto gemm = [&](const bf16_t* A, const bf16_t* B, const float* bias, float* C,
                  int M, int N, int Kd, int act) {
    int waves = (M / 32) * (N / 32);           // one wave per 32x32 output tile
    gemm_bf16_wmma<<<dim3((waves + 3) / 4), dim3(128), 0, stream>>>(A, B, bias, C, M, N, Kd, act);
  };

  // ----- 1. weight conversions to bf16 -----
  cvt_fc_pad_kernel<<<(KPAD * HID + 255) / 256, 256, 0, stream>>>(fc_in_W, WFC);
  for (int i = 0; i < NGCN; ++i)
    cvt_bf16_kernel<<<(HID * HID + 255) / 256, 256, 0, stream>>>(conv_W[i], WCV + (size_t)i * HID * HID, HID * HID);
  cvt_bf16_kernel<<<(HID * HID + 255) / 256, 256, 0, stream>>>(cadj_Wq, WQB, HID * HID);
  cvt_bf16_kernel<<<(HID * HID + 255) / 256, 256, 0, stream>>>(cadj_Wk, WKB, HID * HID);
  cvt_transpose_kernel<<<(HID * HID + 255) / 256, 256, 0, stream>>>(pw1, PW1T, HID, HID);
  cvt_transpose_kernel<<<(HID * HID + 255) / 256, 256, 0, stream>>>(pw2, PW2T, HID, HID);
  cvt_bf16_kernel<<<(DSEQ * HID + 255) / 256, 256, 0, stream>>>(ntp_W, NTPW, DSEQ * HID);
  cvt_bf16_kernel<<<(DSEQ * HID + 255) / 256, 256, 0, stream>>>(ntg_W, NTGW, DSEQ * HID);

  // ----- 2. shape encoder -----
  shape_ln_in_kernel<<<LL / 256, 256, 0, stream>>>(shp, se_ln_in_g, se_ln_in_b, SA);
  for (int lyr = 0; lyr < 2; ++lyr) {
    int bi = 7 + lyr * 16;
    const float* Wq = (const float*)d_in[bi + 0];  const float* bq = (const float*)d_in[bi + 1];
    const float* Wk = (const float*)d_in[bi + 2];  const float* bk = (const float*)d_in[bi + 3];
    const float* Wv = (const float*)d_in[bi + 4];  const float* bv = (const float*)d_in[bi + 5];
    const float* Wo = (const float*)d_in[bi + 6];  const float* bo = (const float*)d_in[bi + 7];
    const float* l1g = (const float*)d_in[bi + 8]; const float* l1b = (const float*)d_in[bi + 9];
    const float* l2g = (const float*)d_in[bi + 10]; const float* l2b = (const float*)d_in[bi + 11];
    const float* W1 = (const float*)d_in[bi + 12]; const float* b1 = (const float*)d_in[bi + 13];
    const float* W2 = (const float*)d_in[bi + 14]; const float* b2 = (const float*)d_in[bi + 15];
    shape_qkv_kernel<<<LL / 256, 256, 0, stream>>>(SA, Wq, bq, Wk, bk, Wv, bv, SQ, SK, SV);
    shape_attn_kernel<<<2 * LL / 256, 256, 0, stream>>>(SQ, SK, SV, SO);
    shape_proj_ln_kernel<<<LL / 256, 256, 0, stream>>>(SA, SO, Wo, bo, l1g, l1b);
    shape_ff_ln_kernel<<<LL / 256, 256, 0, stream>>>(SA, W1, b1, W2, b2, l2g, l2b);
  }
  shape_fnn_ln_kernel<<<LL / 256, 256, 0, stream>>>(SA, fnn_W1, fnn_b1, fnn_W2, fnn_b2,
                                                    se_ln_out_g, se_ln_out_b, SB);

  // ----- 3. align + concat -> XBF -----
  align_concat_kernel<<<LL, 256, 0, stream>>>(seq, nt, SB, align_s_g, align_s_b,
                                              align_nt_g, align_nt_b, align_sh_g, align_sh_b, XBF);

  // ----- 4. adjacency vals (pass 1) -----
  vals0_kernel<<<LL / 256, 256, 0, stream>>>(knn_w, VALS0);

  // ----- 5. GCN pass 1 (h0 -> P0, evolving h -> P1, output H1 -> P4 + HBF) -----
  gemm(XBF, WFC, fc_in_b, P0, LL, HID, KPAD, 1);       // h0 = relu(x@W+b)
  hipMemcpyAsync(P1, P0, (size_t)LL * HID * 4, hipMemcpyDeviceToDevice, stream);
  for (int i = 1; i <= NGCN; ++i) {
    float theta = fminf(1.f, logf(0.5f / (float)i + 1.f));
    spmm_support_kernel<<<LL, 256, 0, stream>>>(knn_idx, VALS0, P1, P0, P2, SBF);
    gemm(SBF, WCV + (size_t)(i - 1) * HID * HID, nullptr, P3, LL, HID, HID, 0);
    gcn_combine_kernel<<<(LL * HID + 255) / 256, 256, 0, stream>>>(theta, P3, P2, P1, LL * HID);
  }
  ln256_kernel<<<LL, 256, 0, stream>>>(P1, post_g, post_b, nullptr, P4, HBF);  // H1

  // ----- 6. content-masked adjacency -----
  gemm(HBF, WQB, nullptr, P2, LL, HID, HID, 0);        // Q
  gemm(HBF, WKB, nullptr, P3, LL, HID, HID, 0);        // K
  sim_kernel<<<(LL * KNN + 255) / 256, 256, 0, stream>>>(P2, P3, knn_idx, SIM);
  vals1_kernel<<<LL / 256, 256, 0, stream>>>(SIM, knn_w, cadj_mix, VALS1);

  // ----- 7. GCN pass 2 (same x => same h0, reuse P0) -----
  hipMemcpyAsync(P1, P0, (size_t)LL * HID * 4, hipMemcpyDeviceToDevice, stream);
  for (int i = 1; i <= NGCN; ++i) {
    float theta = fminf(1.f, logf(0.5f / (float)i + 1.f));
    spmm_support_kernel<<<LL, 256, 0, stream>>>(knn_idx, VALS1, P1, P0, P2, SBF);
    gemm(SBF, WCV + (size_t)(i - 1) * HID * HID, nullptr, P3, LL, HID, HID, 0);
    gcn_combine_kernel<<<(LL * HID + 255) / 256, 256, 0, stream>>>(theta, P3, P2, P1, LL * HID);
  }
  ln256_kernel<<<LL, 256, 0, stream>>>(P1, post_g, post_b, nullptr, P4, nullptr); // H2 -> P4

  // ----- 8. conv refine: H3 = LN(block2(block1(H2))) + H2 -----
  dwconv_kernel<<<(LL * HID + 255) / 256, 256, 0, stream>>>(P4, dw1, HBF, 1);
  gemm(HBF, PW1T, nullptr, P0, LL, HID, HID, 4);       // gelu(pw1)
  dwconv_kernel<<<(LL * HID + 255) / 256, 256, 0, stream>>>(P0, dw2, HBF, 2);
  gemm(HBF, PW2T, nullptr, P1, LL, HID, HID, 4);       // gelu(pw2)
  ln256_kernel<<<LL, 256, 0, stream>>>(P1, cnn_g, cnn_b, P4, P2, nullptr);       // H3 -> P2

  // ----- 9. query-attention readout -> GV -----
  readout_att_kernel<<<(4 * LL + 255) / 256, 256, 0, stream>>>(readout_q, P2, ATT);
  readout_softmax_kernel<<<4, 256, 0, stream>>>(ATT);
  wsum_kernel<<<LL / 256, 256, 0, stream>>>(ATT, WSM);
  gvec_kernel<<<1, 256, 0, stream>>>(WSM, P2, GV);

  // ----- 10. gated nt summary -> VV -----
  nt_ln_kernel<<<LL, 256, 0, stream>>>(nt, ntp_ln_g, ntp_ln_b, ntg_ln_g, ntg_ln_b, NTP, NTG);
  gemm(NTP, NTPW, ntp_b, P0, LL, HID, DSEQ, 2);        // tanh
  gemm(NTG, NTGW, ntg_b, P1, LL, HID, DSEQ, 3);        // sigmoid
  vmean_kernel<<<1, 256, 0, stream>>>(P0, P1, VV);

  // ----- 11. MLP head -> d_out[0] -----
  head_kernel<<<1, 256, 0, stream>>>(GV, VV, h1_W, h1_b, h2_W, h2_b, h3_W, h3_b, (float*)d_out);
}
